// EGATConv_3015067042352
// MI455X (gfx1250) — matrix operand order repaired and checked by reading, not compiled, not run on hardware
//
#include <hip/hip_runtime.h>
#include <hip/hip_bf16.h>

typedef __attribute__((ext_vector_type(16))) _Float16 v16h;
typedef __attribute__((ext_vector_type(8)))  float    v8f;

#define C_IN  256
#define C_OUT 64
#define HEADS 4

// ---------- float <-> orderable unsigned key (for atomicMax on floats) ------
__device__ __forceinline__ unsigned fkey_encode(float f) {
    unsigned b = __float_as_uint(f);
    return (b & 0x80000000u) ? ~b : (b | 0x80000000u);
}
__device__ __forceinline__ float fkey_decode(unsigned k) {
    unsigned b = (k & 0x80000000u) ? (k & 0x7fffffffu) : ~k;
    return __uint_as_float(b);
}

__device__ __forceinline__ void edge_rc(const int* __restrict__ ei, int E, int e,
                                        int& r, int& c) {
    if (e < E) { r = ei[e]; c = ei[E + e]; }
    else       { r = e - E; c = e - E; }
}

// 8-wide f32 -> packed f16 convert + single ds_store_b128
union Pack8 { _Float16 h[8]; uint4 u4; };
__device__ __forceinline__ void cvt_store8(_Float16* dst, const float4 a, const float4 b) {
    Pack8 p;
    p.h[0] = (_Float16)a.x; p.h[1] = (_Float16)a.y;
    p.h[2] = (_Float16)a.z; p.h[3] = (_Float16)a.w;
    p.h[4] = (_Float16)b.x; p.h[5] = (_Float16)b.y;
    p.h[6] = (_Float16)b.z; p.h[7] = (_Float16)b.w;
    *(uint4*)dst = p.u4;
}

// ============================================================================
// Kernel 1: xp = x @ W  via v_wmma_f32_16x16x32_f16
// Block: 128 threads (4 waves). Each block owns 64 output rows (4 subtiles of
// 16); each wave owns a 16-wide N slice of the 64 output columns.
//
// Staging exploits a property of the CDNA5 16-bit A-fragment layout:
// for k-group g (kk = 8g..8g+7 within a 32-wide K block), the fragment
// element indices e are the contiguous run [ (g>>1)*8 .. +7 ] in lane
// (hi<<4)|m with hi = g&1.  So 8 consecutive k's = one ds_store_b128.
// ============================================================================
__global__ void egat_gemm_wmma(const float* __restrict__ x,
                               const float* __restrict__ w,
                               float* __restrict__ xp, int N) {
    __shared__ __attribute__((aligned(32))) _Float16 sW[C_IN * C_OUT]; // 32KB
    __shared__ __attribute__((aligned(32))) _Float16 sA[8 * 32 * 16];  //  8KB

    const int t    = threadIdx.x;
    const int lane = t & 31;
    const int wave = t >> 5;          // 0..3
    const int noff = wave * 16;       // N offset of this wave's tile

    // Stage W (row-major [256][64]) as f16; B-fragment wants lane%16 = N column,
    // halves = 16 K values -> gathered per element from row-major sW, stride 64.
    // 8 elements per step: 2x global_load_b128 + 1x ds_store_b128.
    for (int g = t; g < (C_IN * C_OUT) / 8; g += 128) {
        const float4* src = (const float4*)(w + g * 8);
        cvt_store8(&sW[g * 8], src[0], src[1]);
    }
    __syncthreads();

    const int mblock = blockIdx.x * 64;
    #pragma unroll 1
    for (int sub = 0; sub < 4; ++sub) {
        const int mbase = mblock + sub * 16;
        if (mbase >= N) break;
        __syncthreads();  // previous subtile's reads of sA are done

        // A tile 16x256 -> LDS in fragment order, 8 k's at a time.
        // group index i over 16*256/8 = 512 groups: m = i>>5, g32 = i&31
        // (g32 = kb*4 + g), kb = g32>>2, g = g32&3.
        for (int i = t; i < 512; i += 128) {
            const int m   = i >> 5;
            const int kb  = (i >> 2) & 7;
            const int g   = i & 3;
            const int k   = (kb << 5) + (g << 3);       // first k of group
            const int hi  = g & 1;
            const int eb  = (g >> 1) << 3;              // 0 or 8
            int srow = mbase + m; if (srow >= N) srow = N - 1;  // clamp; stores guarded
            const float4* src = (const float4*)(x + (size_t)srow * C_IN + k);
            _Float16* dst = &sA[((((kb << 5) | (hi << 4) | m) << 4) + eb)];
            cvt_store8(dst, src[0], src[1]);
        }
        __syncthreads();

        v8f acc = {};
        #pragma unroll
        for (int kb = 0; kb < 8; ++kb) {
            v16h a = *(const v16h*)&sA[((kb << 5) | lane) << 4];
            v16h b;
            #pragma unroll
            for (int s = 0; s < 16; ++s) {
                const int kk = (s < 8) ? (((lane >> 4) << 3) + s) : (16 + ((lane >> 4) << 3) + (s - 8));
                b[s] = sW[((kb * 32 + kk) << 6) + noff + (lane & 15)];
            }
            acc = __builtin_amdgcn_wmma_f32_16x16x32_f16(
                false, a, false, b, (short)0, acc, false, false);
        }

        // C/D layout: VGPR r, lane L -> M = r + (L>=16 ? 8 : 0), N = L&15
        const int ncol  = noff + (lane & 15);
        const int rbase = mbase + ((lane >> 4) << 3);
        float* o = xp + (size_t)rbase * C_OUT + ncol;
        if (mbase + 16 <= N) {
            // full tile (uniform branch): unconditional strided stores
            #pragma unroll
            for (int r = 0; r < 8; ++r) o[(size_t)r * C_OUT] = acc[r];
        } else {
            #pragma unroll
            for (int r = 0; r < 8; ++r)
                if (rbase + r < N) o[(size_t)r * C_OUT] = acc[r];
        }
    }
}

// ============================================================================
// Kernel 2: init — zero aggregation output, amax keys (= -inf), denom.
// ============================================================================
__global__ void egat_init(float* __restrict__ out, unsigned* __restrict__ amax,
                          float* __restrict__ denom, int N) {
    const int tid = blockIdx.x * blockDim.x + threadIdx.x;
    if (tid < N * C_OUT) out[tid] = 0.0f;
    if (tid < N * HEADS) { amax[tid] = 0u; denom[tid] = 0.0f; }
}

// ============================================================================
// Kernel 3: per-edge logits + leaky-relu*100, atomic segment-max per (row, h).
// ============================================================================
__global__ void egat_logits(const float* __restrict__ xp,
                            const int* __restrict__ ei,
                            const float* __restrict__ edge_attr,
                            const float* __restrict__ att_w,   // [128][4]
                            const float* __restrict__ att_b,   // [4]
                            float* __restrict__ logits,        // [EN][4]
                            unsigned* __restrict__ amax,       // [N][4]
                            int E, int EN) {
    __shared__ float s_attw[2 * C_OUT * HEADS];   // 512
    __shared__ float s_attb[HEADS];
    for (int i = threadIdx.x; i < 2 * C_OUT * HEADS; i += blockDim.x) s_attw[i] = att_w[i];
    if (threadIdx.x < HEADS) s_attb[threadIdx.x] = att_b[threadIdx.x];
    __syncthreads();

    const int e = blockIdx.x * blockDim.x + threadIdx.x;
    if (e >= EN) return;

    int r, c; edge_rc(ei, E, e, r, c);
    const float ea = (e < E) ? edge_attr[e] : 1.0f;

    const float4* xr = (const float4*)(xp + (size_t)r * C_OUT);
    const float4* xc = (const float4*)(xp + (size_t)c * C_OUT);

    float acc[HEADS];
    #pragma unroll
    for (int h = 0; h < HEADS; ++h) acc[h] = s_attb[h];

    #pragma unroll
    for (int i = 0; i < C_OUT / 4; ++i) {
        const float4 a = xr[i];
        const float4 b = xc[i];
        const int c0 = i * 4;
        const float av[4] = {a.x, a.y, a.z, a.w};
        const float bv[4] = {b.x, b.y, b.z, b.w};
        #pragma unroll
        for (int j = 0; j < 4; ++j) {
            #pragma unroll
            for (int h = 0; h < HEADS; ++h) {
                acc[h] = fmaf(av[j], s_attw[(c0 + j) * HEADS + h], acc[h]);
                acc[h] = fmaf(bv[j], s_attw[(C_OUT + c0 + j) * HEADS + h], acc[h]);
            }
        }
    }

    const float aea = fabsf(ea);
    #pragma unroll
    for (int h = 0; h < HEADS; ++h) {
        float v = acc[h] * aea;
        v = (v > 0.0f ? v : 0.2f * v) * 100.0f;     // leaky_relu * 100
        logits[(size_t)e * HEADS + h] = v;
        atomicMax(&amax[r * HEADS + h], fkey_encode(v));
    }
}

// ============================================================================
// Kernel 4: ex = exp(logit - amax[row]); atomic segment-sum into denom.
// ============================================================================
__global__ void egat_expsum(const int* __restrict__ ei,
                            float* __restrict__ logits,        // in: logit, out: ex
                            const unsigned* __restrict__ amax,
                            float* __restrict__ denom,
                            int E, int EN) {
    const int e = blockIdx.x * blockDim.x + threadIdx.x;
    if (e >= EN) return;
    int r, c; edge_rc(ei, E, e, r, c);
    #pragma unroll
    for (int h = 0; h < HEADS; ++h) {
        const float mx = fkey_decode(amax[r * HEADS + h]);
        const float ex = __expf(logits[(size_t)e * HEADS + h] - mx);
        logits[(size_t)e * HEADS + h] = ex;
        atomicAdd(&denom[r * HEADS + h], ex);
    }
}

// ============================================================================
// Kernel 5: normalize, mean over heads -> alpha output + amean scratch;
// also emit alpha_index (row;col) as floats (values < 2^24, exact).
// ============================================================================
__global__ void egat_alpha(const int* __restrict__ ei,
                           const float* __restrict__ logits,   // ex
                           const float* __restrict__ denom,
                           float* __restrict__ alpha_out,      // [EN]
                           float* __restrict__ amean,          // [EN]
                           float* __restrict__ aidx_out,       // [2][EN]
                           int E, int EN) {
    const int e = blockIdx.x * blockDim.x + threadIdx.x;
    if (e >= EN) return;
    int r, c; edge_rc(ei, E, e, r, c);
    float s = 0.0f;
    #pragma unroll
    for (int h = 0; h < HEADS; ++h)
        s += logits[(size_t)e * HEADS + h] / (denom[r * HEADS + h] + 1e-16f);
    const float a = 0.25f * s;
    alpha_out[e] = a;
    amean[e]     = a;
    aidx_out[e]        = (float)r;
    aidx_out[EN + e]   = (float)c;
}

// ============================================================================
// Kernel 6: out[row] += xp[col] * alpha_mean ; 16 threads per edge, float4 each.
// ============================================================================
__global__ void egat_aggregate(const int* __restrict__ ei,
                               const float* __restrict__ xp,
                               const float* __restrict__ amean,
                               float* __restrict__ out,
                               int E, int EN) {
    const int tid = blockIdx.x * blockDim.x + threadIdx.x;
    const int e = tid >> 4;
    if (e >= EN) return;
    const int q = tid & 15;             // float4 slot 0..15 (64 channels)
    int r, c; edge_rc(ei, E, e, r, c);
    const float a = amean[e];
    const float4 v = ((const float4*)(xp + (size_t)c * C_OUT))[q];
    float* o = out + (size_t)r * C_OUT + q * 4;
    atomicAdd(o + 0, v.x * a);
    atomicAdd(o + 1, v.y * a);
    atomicAdd(o + 2, v.z * a);
    atomicAdd(o + 3, v.w * a);
}

// ============================================================================
extern "C" void kernel_launch(void* const* d_in, const int* in_sizes, int n_in,
                              void* d_out, int out_size, void* d_ws, size_t ws_size,
                              hipStream_t stream) {
    const float* x         = (const float*)d_in[0];
    const int*   edge_idx  = (const int*)  d_in[1];
    const float* edge_attr = (const float*)d_in[2];
    const float* weight    = (const float*)d_in[3];
    const float* att_w     = (const float*)d_in[4];
    const float* att_b     = (const float*)d_in[5];

    const int N  = in_sizes[0] / C_IN;     // 50000
    const int E  = in_sizes[2];            // 800000
    const int EN = E + N;                  // with self loops

    // ---- workspace layout (floats) ----
    float*    ws     = (float*)d_ws;
    float*    xp     = ws;                                  // N*64
    float*    logits = xp + (size_t)N * C_OUT;              // EN*4
    unsigned* amax   = (unsigned*)(logits + (size_t)EN * HEADS); // N*4
    float*    denom  = (float*)(amax + (size_t)N * HEADS);  // N*4
    float*    amean  = denom + (size_t)N * HEADS;           // EN

    // ---- output layout (floats): out | alpha | alpha_index ----
    float* out       = (float*)d_out;                       // N*64
    float* alpha_out = out + (size_t)N * C_OUT;             // EN
    float* aidx_out  = alpha_out + EN;                      // 2*EN

    // 1) xp = x @ W   (WMMA f16 -> f32)
    egat_gemm_wmma<<<(N + 63) / 64, 128, 0, stream>>>(x, weight, xp, N);

    // 2) init accumulators
    {
        const int total = N * C_OUT;
        egat_init<<<(total + 255) / 256, 256, 0, stream>>>(out, amax, denom, N);
    }

    // 3) logits + segment max
    egat_logits<<<(EN + 255) / 256, 256, 0, stream>>>(
        xp, edge_idx, edge_attr, att_w, att_b, logits, amax, E, EN);

    // 4) exp + segment sum
    egat_expsum<<<(EN + 255) / 256, 256, 0, stream>>>(
        edge_idx, logits, amax, denom, E, EN);

    // 5) normalized mean alpha + index outputs
    egat_alpha<<<(EN + 255) / 256, 256, 0, stream>>>(
        edge_idx, logits, denom, alpha_out, amean, aidx_out, E, EN);

    // 6) weighted scatter-aggregate
    {
        const long long total = (long long)EN * 16;
        egat_aggregate<<<(int)((total + 255) / 256), 256, 0, stream>>>(
            edge_idx, xp, amean, out, E, EN);
    }

    (void)n_in; (void)out_size; (void)ws_size;
}
